// DenseQueryAttention_25280177504720
// MI455X (gfx1250) — matrix operand order repaired and checked by reading, not compiled, not run on hardware
//
#include <hip/hip_runtime.h>
#include <hip/hip_bf16.h>

// ---------------------------------------------------------------------------
// Problem constants (from reference): L=1024, B=4, E=1024, H=16, DK=64, TOPK=32
// ---------------------------------------------------------------------------
#define LSEQ 1024
#define NB   4
#define EDIM 1024
#define NH   16
#define DK   64
#define TOPK 32
#define MROWS (LSEQ * NB)   // 4096 GEMM rows

#define AS1 __attribute__((address_space(1)))
#define AS3 __attribute__((address_space(3)))

typedef __attribute__((ext_vector_type(16))) __bf16 v16bf;
typedef __attribute__((ext_vector_type(8)))  __bf16 v8bf;
typedef __attribute__((ext_vector_type(8)))  float  v8f;
typedef __attribute__((ext_vector_type(4)))  int    v4i;

#if defined(__has_builtin)
#  if __has_builtin(__builtin_amdgcn_global_load_async_to_lds_b128)
#    define HAVE_ASYNC_LDS 1
#  else
#    define HAVE_ASYNC_LDS 0
#  endif
#  if __has_builtin(__builtin_amdgcn_s_wait_asynccnt)
#    define WAIT_ASYNC(n) __builtin_amdgcn_s_wait_asynccnt(n)
#  else
#    define WAIT_ASYNC(n) asm volatile("s_wait_asynccnt %0" ::"n"(n))
#  endif
#else
#  define HAVE_ASYNC_LDS 0
#  define WAIT_ASYNC(n)
#endif

// ---------------------------------------------------------------------------
// f32 -> bf16 conversion (grid-stride)
// ---------------------------------------------------------------------------
__global__ __launch_bounds__(256) void cvt_f32_bf16(const float* __restrict__ in,
                                                    __bf16* __restrict__ out, int n) {
    int i = blockIdx.x * blockDim.x + threadIdx.x;
    int stride = gridDim.x * blockDim.x;
    for (; i < n; i += stride) out[i] = (__bf16)in[i];
}

// ---------------------------------------------------------------------------
// LDS tile staging.
// Tile = 128 rows x 32 bf16 (64 B per row), staged with a 40-element (80 B)
// row pitch so fragment reads hit a period-16 bank pattern (no conflicts).
// Copy = 512 chunks of 16 B; 256 threads x 2 chunks each, issued as
// GLOBAL_LOAD_ASYNC_TO_LDS_B128 (ASYNCcnt) when available.
// ---------------------------------------------------------------------------
#define LDS_PITCH 40

__device__ __forceinline__ void tile_to_lds(const __bf16* __restrict__ src, int stride,
                                            int row0, int k, __bf16* dst, int tid) {
#pragma unroll
    for (int c = 0; c < 2; ++c) {
        int chunk = tid * 2 + c;            // 0..511
        int row   = chunk >> 2;             // 4 x 16B chunks per 64B row
        int off   = (chunk & 3) * 8;        // bf16 elements
        const __bf16* gp = src + (size_t)(row0 + row) * stride + k + off;
        __bf16*       sp = dst + row * LDS_PITCH + off;
#if HAVE_ASYNC_LDS
        __builtin_amdgcn_global_load_async_to_lds_b128(
            (AS1 v4i*)(AS1 void*)(void*)gp, (AS3 v4i*)(AS3 void*)sp, 0, 0);
#else
        *(v8bf*)sp = *(const v8bf*)gp;
#endif
    }
}

// ---------------------------------------------------------------------------
// WMMA fragment loaders from LDS, per ISA 7.12.2 VGPR layouts (wave32).
// A (16x32 bf16, MxK): lane 0-15 row=lane holds K[0..7],K[16..23];
//                      lane 16-31 row=lane-16 holds K[8..15],K[24..31]
// B (32x16 bf16, KxN): lane 0-15 col=lane holds K[0..7],K[16..23];
//                      lane 16-31 col=lane-16 holds K[8..15],K[24..31]
// (B's column n == row n of the staged W tile: X @ W.T)
// ---------------------------------------------------------------------------
__device__ __forceinline__ v16bf lds_frag_a(const __bf16* s, int row0, int lane) {
    int r    = row0 + (lane & 15);
    int koff = (lane >> 4) << 3;
    const __bf16* p = s + r * LDS_PITCH + koff;
    v8bf lo = *(const v8bf*)(p);
    v8bf hi = *(const v8bf*)(p + 16);
    return __builtin_shufflevector(lo, hi, 0,1,2,3,4,5,6,7,8,9,10,11,12,13,14,15);
}

__device__ __forceinline__ v16bf lds_frag_b(const __bf16* s, int col0, int lane) {
    int c    = col0 + (lane & 15);
    int koff = (lane >> 4) << 3;
    const __bf16* p = s + c * LDS_PITCH + koff;
    v8bf lo = *(const v8bf*)(p);
    v8bf hi = *(const v8bf*)(p + 16);
    return __builtin_shufflevector(lo, hi, 0,1,2,3,4,5,6,7,8,9,10,11,12,13,14,15);
}

// ---------------------------------------------------------------------------
// Tiled bf16 WMMA GEMM:  out = (A[MxK] @ W[NxK].T + bias) * scale
// Block = 256 threads = 8 waves; block tile 128x128; wave tile 32x64;
// double-buffered async Global->LDS staging of 128x32 A/B slabs.
// mode 0: out row-major f32 [M x N]
// mode 1: out scattered to [B, H, L, DK] f32 (row m = l*NB + b; col n = h*DK + d)
// ---------------------------------------------------------------------------
__global__ __launch_bounds__(256) void gemm_bf16_wmma(const __bf16* __restrict__ A,
                                                      const __bf16* __restrict__ W,
                                                      const float* __restrict__ bias,
                                                      float* __restrict__ out,
                                                      int M, int N, int K,
                                                      float scale, int mode) {
    __shared__ __bf16 sA[2][128 * LDS_PITCH];
    __shared__ __bf16 sB[2][128 * LDS_PITCH];

    const int tid    = threadIdx.x;
    const int lane   = tid & 31;
    const int waveId = tid >> 5;
    const int wm = waveId & 3;   // 4 waves along M
    const int wn = waveId >> 2;  // 2 waves along N
    const int mTile = blockIdx.y * 128;
    const int nTile = blockIdx.x * 128;

    v8f acc[2][4] = {};

    tile_to_lds(A, K, mTile, 0, sA[0], tid);   // 2 async ops
    tile_to_lds(W, K, nTile, 0, sB[0], tid);   // 2 async ops
    int cur = 0;

    for (int k = 0; k < K; k += 32) {
        if (k + 32 < K) {
            tile_to_lds(A, K, mTile, k + 32, sA[cur ^ 1], tid);
            tile_to_lds(W, K, nTile, k + 32, sB[cur ^ 1], tid);
            WAIT_ASYNC(4);   // current stage's 4 ops retired (in-order)
        } else {
            WAIT_ASYNC(0);
        }
        __syncthreads();

        v16bf afrag[2];
        afrag[0] = lds_frag_a(sA[cur], wm * 32,      lane);
        afrag[1] = lds_frag_a(sA[cur], wm * 32 + 16, lane);
        v16bf bfrag[4];
        bfrag[0] = lds_frag_b(sB[cur], wn * 64 +  0, lane);
        bfrag[1] = lds_frag_b(sB[cur], wn * 64 + 16, lane);
        bfrag[2] = lds_frag_b(sB[cur], wn * 64 + 32, lane);
        bfrag[3] = lds_frag_b(sB[cur], wn * 64 + 48, lane);
#pragma unroll
        for (int i = 0; i < 2; ++i)
#pragma unroll
            for (int j = 0; j < 4; ++j)
                acc[i][j] = __builtin_amdgcn_wmma_f32_16x16x32_bf16(
                    false, afrag[i], false, bfrag[j], (short)0, acc[i][j], false, false);

        __syncthreads();
        cur ^= 1;
    }

    // Epilogue: C/D 16x16 f32 layout — lane 0-15: VGPR r -> (M=r, N=lane);
    //           lane 16-31: VGPR r -> (M=r+8, N=lane-16)
    const int lr = lane & 15;
    const int lh = (lane >> 4) * 8;
#pragma unroll
    for (int i = 0; i < 2; ++i) {
#pragma unroll
        for (int j = 0; j < 4; ++j) {
#pragma unroll
            for (int r = 0; r < 8; ++r) {
                int row = mTile + wm * 32 + i * 16 + r + lh;
                int col = nTile + wn * 64 + j * 16 + lr;
                float v = (acc[i][j][r] + bias[col]) * scale;
                if (mode == 0) {
                    out[(size_t)row * N + col] = v;
                } else {
                    int l = row >> 2, bb = row & (NB - 1);
                    int h = col >> 6, d  = col & (DK - 1);
                    out[(((size_t)(bb * NH + h)) * LSEQ + l) * DK + d] = v;
                }
            }
        }
    }
}

// ---------------------------------------------------------------------------
// Sparse top-k attention: 1 wave (32 lanes) per (b,h,l) query; lane = topk slot
// ---------------------------------------------------------------------------
__global__ __launch_bounds__(128) void attn_topk_kernel(const float* __restrict__ qbuf,
                                                        const float* __restrict__ kbuf,
                                                        const float* __restrict__ vbuf,
                                                        const int* __restrict__ mask,
                                                        __bf16* __restrict__ ctx,
                                                        float* __restrict__ attn_full) {
    __shared__ float s_attn[4][TOPK];
    __shared__ int   s_idx[4][TOPK];

    const int gtid = blockIdx.x * 128 + threadIdx.x;
    const int wave = gtid >> 5;
    const int lane = threadIdx.x & 31;
    const int wib  = threadIdx.x >> 5;

    const int l  = wave % LSEQ;
    const int h  = (wave / LSEQ) % NH;
    const int bb = wave / (LSEQ * NH);
    const int bh = bb * NH + h;

    const float* qrow  = qbuf + ((size_t)bh * LSEQ + l) * DK;
    const float* kbase = kbuf + (size_t)bh * LSEQ * DK;
    const float* vbase = vbuf + (size_t)bh * LSEQ * DK;

    const int idx = mask[((size_t)bh * LSEQ + l) * TOPK + lane];

    // per-lane 64-dim dot product (q is wave-uniform -> broadcast loads)
    const float4* q4 = (const float4*)qrow;
    const float4* k4 = (const float4*)(kbase + (size_t)idx * DK);
    float s = 0.f;
#pragma unroll
    for (int d4 = 0; d4 < DK / 4; ++d4) {
        float4 qv = q4[d4], kv = k4[d4];
        s += qv.x * kv.x + qv.y * kv.y + qv.z * kv.z + qv.w * kv.w;
    }

    // softmax across the 32 lanes (wave32)
    float mx = s;
#pragma unroll
    for (int off = 16; off >= 1; off >>= 1) mx = fmaxf(mx, __shfl_xor(mx, off, 32));
    float e = __expf(s - mx);
    float sum = e;
#pragma unroll
    for (int off = 16; off >= 1; off >>= 1) sum += __shfl_xor(sum, off, 32);
    float a = e / sum;

    attn_full[((size_t)bh * LSEQ + l) * TOPK + lane] = a;
    s_attn[wib][lane] = a;
    s_idx[wib][lane]  = idx;
    __syncthreads();

    // d-parallel weighted sum: each lane owns 2 of the 64 dims
    float acc0 = 0.f, acc1 = 0.f;
#pragma unroll 8
    for (int t = 0; t < TOPK; ++t) {
        float at = s_attn[wib][t];
        const float2* v2 = (const float2*)(vbase + (size_t)s_idx[wib][t] * DK);
        float2 vv = v2[lane];
        acc0 += at * vv.x;
        acc1 += at * vv.y;
    }
    size_t cbase = ((size_t)l * NB + bb) * EDIM + h * DK + lane * 2;
    ctx[cbase]     = (__bf16)acc0;
    ctx[cbase + 1] = (__bf16)acc1;
}

// ---------------------------------------------------------------------------
// attn.sum(axis=heads) / H  ->  [B, L, TOPK]  (deterministic, no atomics)
// ---------------------------------------------------------------------------
__global__ __launch_bounds__(256) void attn_head_reduce(const float* __restrict__ attn_full,
                                                        float* __restrict__ out, int n) {
    int i = blockIdx.x * blockDim.x + threadIdx.x;
    if (i >= n) return;
    int t  = i % TOPK;
    int l  = (i / TOPK) % LSEQ;
    int bb = i / (TOPK * LSEQ);
    float s = 0.f;
#pragma unroll
    for (int h = 0; h < NH; ++h)
        s += attn_full[(((size_t)(bb * NH + h)) * LSEQ + l) * TOPK + t];
    out[i] = s * (1.0f / NH);
}

// ---------------------------------------------------------------------------
extern "C" void kernel_launch(void* const* d_in, const int* in_sizes, int n_in,
                              void* d_out, int out_size, void* d_ws, size_t ws_size,
                              hipStream_t stream) {
    (void)in_sizes; (void)n_in; (void)out_size; (void)ws_size;

    const float* query = (const float*)d_in[0];
    const float* key   = (const float*)d_in[1];
    const float* value = (const float*)d_in[2];
    const int*   mask  = (const int*)  d_in[3];
    const float* w_in  = (const float*)d_in[4];
    const float* b_in  = (const float*)d_in[5];
    const float* w_out = (const float*)d_in[6];
    const float* b_out = (const float*)d_in[7];
    float* out_main = (float*)d_out;                               // [L,B,E]
    float* out_attn = out_main + (size_t)LSEQ * NB * EDIM;         // [B,L,TOPK]

    // workspace carve-up (256B aligned)
    char*  ws = (char*)d_ws;
    size_t o  = 0;
    auto carve = [&](size_t bytes) -> void* {
        void* p = ws + o;
        o += (bytes + 255) & ~(size_t)255;
        return p;
    };
    const size_t nX = (size_t)MROWS * EDIM;
    __bf16* xq_bf  = (__bf16*)carve(nX * 2);
    __bf16* xk_bf  = (__bf16*)carve(nX * 2);
    __bf16* xv_bf  = (__bf16*)carve(nX * 2);
    __bf16* win_bf = (__bf16*)carve((size_t)3 * EDIM * EDIM * 2);
    __bf16* wout_bf= (__bf16*)carve((size_t)EDIM * EDIM * 2);
    float*  qbuf   = (float*) carve(nX * 4);
    float*  kbuf   = (float*) carve(nX * 4);
    float*  vbuf   = (float*) carve(nX * 4);
    __bf16* ctx_bf = (__bf16*)carve(nX * 2);
    float*  attnws = (float*) carve((size_t)NB * NH * LSEQ * TOPK * 4);

    // 1) f32 -> bf16 conversions
    cvt_f32_bf16<<<2048, 256, 0, stream>>>(query, xq_bf, (int)nX);
    cvt_f32_bf16<<<2048, 256, 0, stream>>>(key,   xk_bf, (int)nX);
    cvt_f32_bf16<<<2048, 256, 0, stream>>>(value, xv_bf, (int)nX);
    cvt_f32_bf16<<<2048, 256, 0, stream>>>(w_in,  win_bf, 3 * EDIM * EDIM);
    cvt_f32_bf16<<<2048, 256, 0, stream>>>(w_out, wout_bf, EDIM * EDIM);

    // 2) QKV projections (WMMA bf16), scattered to [B,H,L,DK]; q scaled by DK^-0.5
    dim3 ggrid(EDIM / 128, MROWS / 128);
    const float qscale = 0.125f;  // 64^-0.5
    gemm_bf16_wmma<<<ggrid, 256, 0, stream>>>(xq_bf, win_bf,
                                              b_in, qbuf, MROWS, EDIM, EDIM, qscale, 1);
    gemm_bf16_wmma<<<ggrid, 256, 0, stream>>>(xk_bf, win_bf + (size_t)EDIM * EDIM,
                                              b_in + EDIM, kbuf, MROWS, EDIM, EDIM, 1.0f, 1);
    gemm_bf16_wmma<<<ggrid, 256, 0, stream>>>(xv_bf, win_bf + (size_t)2 * EDIM * EDIM,
                                              b_in + 2 * EDIM, vbuf, MROWS, EDIM, EDIM, 1.0f, 1);

    // 3) top-k attention (1 wave per query), writes bf16 ctx + raw attn
    attn_topk_kernel<<<(NB * NH * LSEQ) / 4, 128, 0, stream>>>(qbuf, kbuf, vbuf, mask,
                                                               ctx_bf, attnws);

    // 4) head-averaged attention output
    int n2 = NB * LSEQ * TOPK;
    attn_head_reduce<<<(n2 + 255) / 256, 256, 0, stream>>>(attnws, out_attn, n2);

    // 5) output projection (WMMA bf16), row-major f32 into d_out
    gemm_bf16_wmma<<<ggrid, 256, 0, stream>>>(ctx_bf, wout_bf, b_out, out_main,
                                              MROWS, EDIM, EDIM, 1.0f, 0);
}
